// _X_65214783423134
// MI455X (gfx1250) — compile-verified, matrix operand order checked by reading, not run on hardware
//
#include <hip/hip_runtime.h>
#include <hip/hip_bf16.h>

// ---------------- problem constants ----------------
#define Dc 512
#define Hc 1024
#define Rc 32
#define Sc 16
#define Bc 4
#define Lc 2048
#define BLc (Bc * Lc)   // 8192

typedef __attribute__((ext_vector_type(16))) __bf16 v16bf;
typedef __attribute__((ext_vector_type(8)))  float  v8f;
typedef __attribute__((ext_vector_type(4)))  unsigned int v4u;
typedef __attribute__((ext_vector_type(8)))  int v8i;
typedef __attribute__((ext_vector_type(4)))  int v4i;

union TileU { v16bf v; uint4 q[2]; };

__device__ __forceinline__ __bf16 f2bf(float f) {
    unsigned int u = __builtin_bit_cast(unsigned int, f);
    u += 0x7fffu + ((u >> 16) & 1u);           // round-to-nearest-even
    unsigned short h = (unsigned short)(u >> 16);
    return __builtin_bit_cast(__bf16, h);
}

__device__ __forceinline__ float sigmoidf_(float x) { return 1.0f / (1.0f + __expf(-x)); }
__device__ __forceinline__ float softplusf_(float x) { return (x > 20.0f) ? x : log1pf(__expf(x)); }

// Load a 16x32 bf16 operand tile (A layout; also used for W/B since B's
// per-lane-column layout == W[N,K] row-major rows).  lanes 0-15: K 0..7 then
// 16..23; lanes 16-31: K 8..15 then 24..31 (ISA 7.12.2).
__device__ __forceinline__ v16bf load_tile16x32(const __bf16* __restrict__ base,
                                                int ld, int row0, int k0, int lane) {
    int r = lane & 15, g = lane >> 4;
    const __bf16* p = base + (size_t)(row0 + r) * (size_t)ld + (size_t)(k0 + g * 8);
    TileU t;
    t.q[0] = *(const uint4*)(p);
    t.q[1] = *(const uint4*)(p + 16);
    return t.v;
}

// ---------------- Tensor Data Mover: 2D tile (rows x kl bf16) -> LDS ----------
// D# packing per CDNA5 ISA ch.8: group0 {count, lds_addr, global_addr, type=2},
// group1 {data_size=1(2B), tensor_dim0=kl, tensor_dim1=rows, tile_dim0=kl,
// tile_dim1=rows, tensor_dim0_stride=strideElems}.  Groups 2/3 zero (2D tile).
__device__ __forceinline__ void tdm_load_2d(unsigned int lds_off, const void* gptr,
                                            unsigned int kl, unsigned int rows,
                                            unsigned int strideElems) {
    unsigned long long ga = (unsigned long long)(uintptr_t)gptr;
    v4u g0;
    g0[0] = 1u;                                            // count=1, user mode
    g0[1] = lds_off;                                       // lds_addr (bytes)
    g0[2] = (unsigned int)ga;                              // global_addr[31:0]
    g0[3] = (unsigned int)((ga >> 32) & 0x1FFFFFFu) | (2u << 30); // addr[56:32] | type=2
    v8i g1;
    g1[0] = (int)(1u << 16);                               // data_size = 2 bytes
    g1[1] = (int)((kl & 0xFFFFu) << 16);                   // tensor_dim0[15:0]
    g1[2] = (int)(((kl >> 16) & 0xFFFFu) | ((rows & 0xFFFFu) << 16)); // dim0 hi | dim1 lo
    g1[3] = (int)(((rows >> 16) & 0xFFFFu) | ((kl & 0xFFFFu) << 16)); // dim1 hi | tile_dim0
    g1[4] = (int)(rows & 0xFFFFu);                         // tile_dim1 (tile_dim2 = 0)
    g1[5] = (int)strideElems;                              // tensor_dim0_stride[31:0]
    g1[6] = 0;                                             // stride hi / dim1_stride lo
    g1[7] = 0;
    v4i g2 = {0, 0, 0, 0};
    v4i g3 = {0, 0, 0, 0};
    v8i g4 = {0, 0, 0, 0, 0, 0, 0, 0};                     // extra group (clang-23 6-arg form)
    __builtin_amdgcn_tensor_load_to_lds(g0, g1, g2, g3, g4, 0);
}

// ---------------- bf16 WMMA GEMM with TDM-staged weight panel ----------------
// C[M,N] = act(A[M,K] * W[N,K]^T + bias), fp32 out; optional bf16 side-output
// of the first bfCols columns.  128 threads = 4 waves (2x2); wave tile =
// 32 x (16*WNT); block tile = 64 x (32*WNT).  B panel staged to LDS in
// 256-K-element chunks by the Tensor Data Mover (one slice per wave).
template<int WNT>
__global__ __launch_bounds__(128)
void k_gemm_tdm(const __bf16* __restrict__ A, const __bf16* __restrict__ W,
                const float* __restrict__ bias, float* __restrict__ C,
                __bf16* __restrict__ Cbf, int M, int N, int K,
                int act, int bfCols)
{
    extern __shared__ __bf16 smemB[];   // [32*WNT rows][min(K,256)]
    const int lane = threadIdx.x & 31;
    const int wave = threadIdx.x >> 5;
    const int wr = wave >> 1, wc = wave & 1;
    const int m0 = blockIdx.y * 64 + wr * 32;
    const int bn = blockIdx.x * (32 * WNT);
    const int nloc0 = wc * (16 * WNT);
    const int sliceRows = (32 * WNT) / 4;
    const int wuni = __builtin_amdgcn_readfirstlane(wave);
    const unsigned int ldsBase = (unsigned int)(uintptr_t)(void*)smemB;

    v8f acc0[WNT] = {};
    v8f acc1[WNT] = {};

    for (int kc = 0; kc < K; kc += 256) {
        const int kl = (K - kc) < 256 ? (K - kc) : 256;
        {   // each wave DMAs its quarter of the B panel (TDM ignores EXEC)
            const int rowG = bn + wuni * sliceRows;
            const __bf16* gsrc = W + (size_t)rowG * (size_t)K + (size_t)kc;
            tdm_load_2d(ldsBase + (unsigned int)(wuni * sliceRows * kl * 2),
                        gsrc, (unsigned int)kl, (unsigned int)sliceRows,
                        (unsigned int)K);
        }
        __builtin_amdgcn_s_wait_tensorcnt(0);
        __syncthreads();

        for (int k0 = 0; k0 < kl; k0 += 32) {
            v16bf a0 = load_tile16x32(A, K, m0,      kc + k0, lane);
            v16bf a1 = load_tile16x32(A, K, m0 + 16, kc + k0, lane);
            #pragma unroll
            for (int tn = 0; tn < WNT; ++tn) {
                v16bf b = load_tile16x32(smemB, kl, nloc0 + tn * 16, k0, lane);
                acc0[tn] = __builtin_amdgcn_wmma_f32_16x16x32_bf16(false, a0, false, b, (short)0, acc0[tn], false, false);
                acc1[tn] = __builtin_amdgcn_wmma_f32_16x16x32_bf16(false, a1, false, b, (short)0, acc1[tn], false, false);
            }
        }
        __syncthreads();   // all waves done reading before next chunk's DMA
    }

    // C/D layout: lane -> N = lane&15 (lane>>4 selects M+8), vgpr v -> M = v
    const int cn = lane & 15;
    const int mg = (lane >> 4) * 8;
    #pragma unroll
    for (int tn = 0; tn < WNT; ++tn) {
        const int nn = bn + nloc0 + tn * 16 + cn;
        const float bval = bias ? bias[nn] : 0.0f;
        #pragma unroll
        for (int half = 0; half < 2; ++half) {
            const v8f accv = half ? acc1[tn] : acc0[tn];
            #pragma unroll
            for (int v = 0; v < 8; ++v) {
                const int mm = m0 + half * 16 + mg + v;
                float xv = accv[v] + bval;
                if (act == 1) xv = softplusf_(xv);
                C[(size_t)mm * (size_t)N + (size_t)nn] = xv;
                if (Cbf != nullptr && nn < bfCols)
                    Cbf[(size_t)mm * (size_t)bfCols + (size_t)nn] = f2bf(xv);
            }
        }
    }
}

// ---------------- fp32 -> bf16 convert ----------------
__global__ void k_f32_to_bf16(const float* __restrict__ src, __bf16* __restrict__ dst, int n) {
    int i = blockIdx.x * blockDim.x + threadIdx.x;
    if (i < n) dst[i] = f2bf(src[i]);
}

// ---------------- LayerNorm(ctx) -> bf16 ----------------
__global__ __launch_bounds__(128)
void k_layernorm_bf16(const float* __restrict__ x, const float* __restrict__ w,
                      const float* __restrict__ b, __bf16* __restrict__ out)
{
    __shared__ float s1s[4], s2s[4];
    const int row = blockIdx.x;
    const float* xr = x + (size_t)row * Dc;
    float vals[4];
    #pragma unroll
    for (int j = 0; j < 4; ++j) vals[j] = xr[threadIdx.x * 4 + j];
    float s1 = 0.f, s2 = 0.f;
    #pragma unroll
    for (int j = 0; j < 4; ++j) { s1 += vals[j]; s2 += vals[j] * vals[j]; }
    #pragma unroll
    for (int off = 16; off; off >>= 1) {
        s1 += __shfl_xor(s1, off, 32);
        s2 += __shfl_xor(s2, off, 32);
    }
    const int wv = threadIdx.x >> 5;
    if ((threadIdx.x & 31) == 0) { s1s[wv] = s1; s2s[wv] = s2; }
    __syncthreads();
    s1 = s1s[0] + s1s[1] + s1s[2] + s1s[3];
    s2 = s2s[0] + s2s[1] + s2s[2] + s2s[3];
    const float mean = s1 * (1.0f / Dc);
    const float var  = s2 * (1.0f / Dc) - mean * mean;
    const float inv  = rsqrtf(var + 1e-5f);
    __bf16* outr = out + (size_t)row * Dc;
    #pragma unroll
    for (int j = 0; j < 4; ++j) {
        const int i = threadIdx.x * 4 + j;
        outr[i] = f2bf((vals[j] - mean) * inv * w[i] + b[i]);
    }
}

// ---------------- gating + causal depthwise conv(k=3) + SiLU ----------------
#define CONV_CHUNK 128
__global__ __launch_bounds__(256)
void k_gate_conv(const float* __restrict__ z, const float* __restrict__ pq,
                 const float* __restrict__ conv_w, const float* __restrict__ conv_b,
                 float* __restrict__ u_f32, __bf16* __restrict__ u_bf16,
                 float* __restrict__ v_f32)
{
    const int h  = blockIdx.y * 256 + threadIdx.x;
    const int b  = blockIdx.z;
    const int l0 = blockIdx.x * CONV_CHUNK;
    const float w0 = conv_w[h * 3 + 0];
    const float w1 = conv_w[h * 3 + 1];
    const float w2 = conv_w[h * 3 + 2];
    const float cb = conv_b[h];

    auto ug = [&](int l) -> float {
        if (l < 0) return 0.0f;
        const size_t idx = ((size_t)(b * Lc + l)) * (size_t)(2 * Hc) + (size_t)h;
        return z[idx] * (1.0f + sigmoidf_(pq[idx]));
    };

    float um2 = ug(l0 - 2);
    float um1 = ug(l0 - 1);
    for (int l = l0; l < l0 + CONV_CHUNK; ++l) {
        const float uc   = ug(l);
        const float conv = w0 * um2 + w1 * um1 + w2 * uc + cb;
        const float sl   = conv * sigmoidf_(conv);           // silu
        const size_t o   = ((size_t)(b * Lc + l)) * (size_t)Hc + (size_t)h;
        u_f32[o]  = sl;
        u_bf16[o] = f2bf(sl);
        const size_t iv = ((size_t)(b * Lc + l)) * (size_t)(2 * Hc) + (size_t)(Hc + h);
        v_f32[o] = z[iv] + pq[iv];
        um2 = um1; um1 = uc;
    }
}

// ---------------- selective scan (sequential in L) ----------------
__global__ __launch_bounds__(256)
void k_scan(const float* __restrict__ dt, const float* __restrict__ u,
            const float* __restrict__ zc, const float* __restrict__ v,
            const float* __restrict__ f_log, const float* __restrict__ g,
            __bf16* __restrict__ ybf)
{
    const int tid  = threadIdx.x;
    const int s    = tid & 15;
    const int hloc = tid >> 4;
    const int h    = blockIdx.y * 16 + hloc;
    const int b    = blockIdx.x;
    const float a  = -__expf(f_log[h * Sc + s]);
    const float gh = g[h];
    float state = 0.0f;
    const size_t base = (size_t)b * Lc;
    for (int l = 0; l < Lc; ++l) {
        const size_t row = base + (size_t)l;
        const float dtv = dt[row * Hc + h];
        const float uv  = u[row * Hc + h];
        const float bs  = zc[row * 64 + 32 + s];
        const float cs  = zc[row * 64 + 48 + s];
        state = __expf(dtv * a) * state + dtv * uv * bs;
        float y = state * cs;
        y += __shfl_xor(y, 1, 32);
        y += __shfl_xor(y, 2, 32);
        y += __shfl_xor(y, 4, 32);
        y += __shfl_xor(y, 8, 32);
        if (s == 0) {
            const float vv = v[row * Hc + h];
            ybf[row * Hc + h] = f2bf((y + uv * gh) * (vv * sigmoidf_(vv)));
        }
    }
}

// ---------------- host launch ----------------
static inline size_t align_up256(size_t x) { return (x + 255) & ~(size_t)255; }

extern "C" void kernel_launch(void* const* d_in, const int* in_sizes, int n_in,
                              void* d_out, int out_size, void* d_ws, size_t ws_size,
                              hipStream_t stream) {
    const float* x      = (const float*)d_in[0];
    const float* ctx    = (const float*)d_in[1];
    const float* Wa     = (const float*)d_in[2];
    const float* ba     = (const float*)d_in[3];
    const float* conv_w = (const float*)d_in[4];
    const float* conv_b = (const float*)d_in[5];
    const float* Wc     = (const float*)d_in[6];
    const float* We     = (const float*)d_in[7];
    const float* be     = (const float*)d_in[8];
    const float* f_log  = (const float*)d_in[9];
    const float* g      = (const float*)d_in[10];
    const float* Wi     = (const float*)d_in[11];
    const float* bi     = (const float*)d_in[12];
    const float* ln_w   = (const float*)d_in[13];
    const float* ln_b   = (const float*)d_in[14];
    const float* Wgb    = (const float*)d_in[15];
    const float* bgb    = (const float*)d_in[16];
    const float* Wgc    = (const float*)d_in[17];
    const float* bgc    = (const float*)d_in[18];
    float* out = (float*)d_out;

    char* ws = (char*)d_ws;
    size_t off = 0;
    auto walloc = [&](size_t bytes) -> void* {
        void* p = ws + off;
        off = align_up256(off + bytes);
        return p;
    };

    __bf16* x_bf    = (__bf16*)walloc((size_t)BLc * Dc * 2);
    __bf16* cln_bf  = (__bf16*)walloc((size_t)BLc * Dc * 2);
    __bf16* Wa_bf   = (__bf16*)walloc((size_t)2 * Hc * Dc * 2);
    __bf16* Wg_bf   = (__bf16*)walloc((size_t)2 * Hc * Dc * 2);
    __bf16* Wc_bf   = (__bf16*)walloc((size_t)64 * Hc * 2);
    __bf16* We_bf   = (__bf16*)walloc((size_t)Hc * Rc * 2);
    __bf16* Wi_bf   = (__bf16*)walloc((size_t)Dc * Hc * 2);
    float*  bg_pack = (float*)walloc((size_t)2 * Hc * 4);
    float*  z       = (float*)walloc((size_t)BLc * 2 * Hc * 4);
    float*  pq      = (float*)walloc((size_t)BLc * 2 * Hc * 4);
    float*  u_f32   = (float*)walloc((size_t)BLc * Hc * 4);
    __bf16* u_bf    = (__bf16*)walloc((size_t)BLc * Hc * 2);
    float*  v_f32   = (float*)walloc((size_t)BLc * Hc * 4);
    float*  zc      = (float*)walloc((size_t)BLc * 64 * 4);
    __bf16* dtraw_bf= (__bf16*)walloc((size_t)BLc * Rc * 2);
    float*  dt      = (float*)walloc((size_t)BLc * Hc * 4);
    __bf16* y_bf    = (__bf16*)walloc((size_t)BLc * Hc * 2);
    (void)ws_size; (void)in_sizes; (void)n_in; (void)out_size;

    auto cvt = [&](const float* src, __bf16* dst, int n) {
        k_f32_to_bf16<<<(n + 255) / 256, 256, 0, stream>>>(src, dst, n);
    };
    auto shbytes = [](int WNT, int K) -> size_t {
        int kc = K < 256 ? K : 256;
        return (size_t)(32 * WNT) * (size_t)kc * 2;
    };

    // stage 0: precision conversion + LN + gate-weight packing
    cvt(x,   x_bf,  BLc * Dc);
    cvt(Wa,  Wa_bf, 2 * Hc * Dc);
    cvt(Wgb, Wg_bf, Hc * Dc);
    cvt(Wgc, Wg_bf + (size_t)Hc * Dc, Hc * Dc);
    cvt(Wc,  Wc_bf, 64 * Hc);
    cvt(We,  We_bf, Hc * Rc);
    cvt(Wi,  Wi_bf, Dc * Hc);
    (void)hipMemcpyAsync(bg_pack,      bgb, Hc * sizeof(float), hipMemcpyDeviceToDevice, stream);
    (void)hipMemcpyAsync(bg_pack + Hc, bgc, Hc * sizeof(float), hipMemcpyDeviceToDevice, stream);
    k_layernorm_bf16<<<BLc, 128, 0, stream>>>(ctx, ln_w, ln_b, cln_bf);

    // stage 1: big WMMA GEMMs (M=8192), block tile 64x128
    {   // z = x @ Wa^T + ba   [BL, 2H]
        dim3 grid((2 * Hc) / 128, BLc / 64);
        k_gemm_tdm<4><<<grid, 128, shbytes(4, Dc), stream>>>(x_bf, Wa_bf, ba, z, nullptr,
                                                             BLc, 2 * Hc, Dc, 0, 0);
    }
    {   // [p|q] = cln @ [Wgb;Wgc]^T + [bgb;bgc]   [BL, 2H]
        dim3 grid((2 * Hc) / 128, BLc / 64);
        k_gemm_tdm<4><<<grid, 128, shbytes(4, Dc), stream>>>(cln_bf, Wg_bf, bg_pack, pq, nullptr,
                                                             BLc, 2 * Hc, Dc, 0, 0);
    }

    // stage 2: gating + causal conv + silu  -> u (f32 + bf16), v (f32)
    {
        dim3 grid(Lc / CONV_CHUNK, Hc / 256, Bc);
        k_gate_conv<<<grid, 256, 0, stream>>>(z, pq, conv_w, conv_b,
                                              u_f32, u_bf, v_f32);
    }

    // stage 3: SSM parameter GEMMs
    {   // zc = u @ Wc^T  [BL, 64]; bf16 copy of dt_raw cols (0..31)
        dim3 grid(64 / 64, BLc / 64);
        k_gemm_tdm<2><<<grid, 128, shbytes(2, Hc), stream>>>(u_bf, Wc_bf, nullptr, zc, dtraw_bf,
                                                             BLc, 64, Hc, 0, Rc);
    }
    {   // dt = softplus(dt_raw @ We^T + be)  [BL, H]
        dim3 grid(Hc / 128, BLc / 64);
        k_gemm_tdm<4><<<grid, 128, shbytes(4, Rc), stream>>>(dtraw_bf, We_bf, be, dt, nullptr,
                                                             BLc, Hc, Rc, 1, 0);
    }

    // stage 4: selective scan (+ u*g, * silu(v)) -> y bf16
    {
        dim3 grid(Bc, Hc / 16);
        k_scan<<<grid, 256, 0, stream>>>(dt, u_f32, zc, v_f32, f_log, g, y_bf);
    }

    // stage 5: out = y @ Wi^T + bi  [BL, D] -> d_out
    {
        dim3 grid(Dc / 64, BLc / 64);
        k_gemm_tdm<2><<<grid, 128, shbytes(2, Hc), stream>>>(y_bf, Wi_bf, bi, out, nullptr,
                                                             BLc, Dc, Hc, 0, 0);
    }
}
